// MultiheadAttention_60035052863605
// MI455X (gfx1250) — compile-verified, matrix operand order checked by reading, not run on hardware
//
#include <hip/hip_runtime.h>
#include <cstdint>

// ---------------------------------------------------------------------------
// MI455X (gfx1250, CDNA5, wave32) multi-head attention forward.
// GEMMs on v_wmma_f32_16x16x32_bf16 (fp32 accumulate).
// GEMM tile staging: Tensor Data Mover (TENSORcnt) when available,
// else async global->LDS (ASYNCcnt), else sync loads. Attention staging
// uses the async global->LDS pipe.
// ---------------------------------------------------------------------------

typedef __attribute__((ext_vector_type(8)))  __bf16 v8bf;
typedef __attribute__((ext_vector_type(16))) __bf16 v16bf;
typedef __attribute__((ext_vector_type(8)))  float  v8f;
typedef __attribute__((ext_vector_type(4)))  int    v4i;
typedef __attribute__((ext_vector_type(8)))  int    v8i;
typedef __attribute__((ext_vector_type(4)))  unsigned int v4u;

#define B_   2
#define L_   2048
#define D_   1024
#define H_   16
#define DH_  64
#define M_   (B_ * L_)
#define SCALE_ 0.03125f         // 1/sqrt(1024)

#define AS1 __attribute__((address_space(1)))
#define AS3 __attribute__((address_space(3)))

#if __has_builtin(__builtin_amdgcn_global_load_async_to_lds_b128)
#define HAVE_ASYNC_LDS 1
#else
#define HAVE_ASYNC_LDS 0
#endif

#if __has_builtin(__builtin_amdgcn_tensor_load_to_lds)
#define HAVE_TDM 1
#else
#define HAVE_TDM 0
#endif

#if HAVE_ASYNC_LDS
static __device__ __forceinline__ void async_ld_b128(void* lds, const void* g) {
  __builtin_amdgcn_global_load_async_to_lds_b128(
      (AS1 v4i*)(uintptr_t)g, (AS3 v4i*)(uintptr_t)lds, 0, 0);
}
static __device__ __forceinline__ void wait_async_le4() {
#if __has_builtin(__builtin_amdgcn_s_wait_asynccnt)
  __builtin_amdgcn_s_wait_asynccnt(4);
#else
  asm volatile("s_wait_asynccnt 0x4" ::: "memory");
#endif
}
static __device__ __forceinline__ void wait_async_0() {
#if __has_builtin(__builtin_amdgcn_s_wait_asynccnt)
  __builtin_amdgcn_s_wait_asynccnt(0);
#else
  asm volatile("s_wait_asynccnt 0x0" ::: "memory");
#endif
}
#endif

#if HAVE_TDM
static __device__ __forceinline__ void wait_tensor_le2() {
#if __has_builtin(__builtin_amdgcn_s_wait_tensorcnt)
  __builtin_amdgcn_s_wait_tensorcnt(2);
#else
  asm volatile("s_wait_tensorcnt 0x2" ::: "memory");
#endif
}
static __device__ __forceinline__ void wait_tensor_0() {
#if __has_builtin(__builtin_amdgcn_s_wait_tensorcnt)
  __builtin_amdgcn_s_wait_tensorcnt(0);
#else
  asm volatile("s_wait_tensorcnt 0x0" ::: "memory");
#endif
}

// TDM 2D tile load: tile_d1 rows x tile_d0 elems (bf16) out of a row-major
// tensor with row stride stride0 elems.  D# per CDNA5 ISA ch.8 (§8.3/8.4).
static __device__ __forceinline__ void
tdm_load_2d_bf16(const void* gaddr, unsigned int lds_byte_addr,
                 unsigned int tensor_d0, unsigned int tensor_d1,
                 unsigned int stride0, unsigned int tile_d0,
                 unsigned int tile_d1) {
  unsigned long long ga = (unsigned long long)(uintptr_t)gaddr;
  v4u g0;
  g0[0] = 1u;                                  // count=1 (valid), user mode
  g0[1] = lds_byte_addr;                       // lds_addr
  g0[2] = (unsigned int)ga;                    // global_addr[31:0]
  g0[3] = (unsigned int)(ga >> 32) | (2u << 30);  // global_addr[56:32] | type=2
  v8i g1;
  g1[0] = (int)(1u << 16);                     // workgroup_mask=0, data_size=2B
  g1[1] = (int)((tensor_d0 & 0xFFFFu) << 16);  // tensor_dim0[15:0]
  g1[2] = (int)(((tensor_d0 >> 16) & 0xFFFFu) |
                ((tensor_d1 & 0xFFFFu) << 16));
  g1[3] = (int)(((tensor_d1 >> 16) & 0xFFFFu) | (tile_d0 << 16));
  g1[4] = (int)tile_d1;                        // tile_dim1 | tile_dim2=0
  g1[5] = (int)stride0;                        // tensor_dim0_stride[31:0]
  g1[6] = 0;                                   // stride0 hi | stride1 lo
  g1[7] = 0;                                   // stride1 hi
  v4i z4 = {0, 0, 0, 0};
  v8i z8 = {0, 0, 0, 0, 0, 0, 0, 0};
  __builtin_amdgcn_tensor_load_to_lds(g0, g1, z4, z4, z8, 0);
}
#endif

static __device__ __forceinline__ v8f wmma_bf16(v16bf a, v16bf b, v8f c) {
  return __builtin_amdgcn_wmma_f32_16x16x32_bf16(false, a, false, b,
                                                 (short)0, c, false, false);
}

// Per-lane 16-bit A/B fragment: two contiguous 8-elem chunks at k=h*8, 16+h*8.
static __device__ __forceinline__ v16bf ldfrag(const __bf16* p) {
  v8bf lo = *(const v8bf*)p;
  v8bf hi = *(const v8bf*)(p + 16);
  return __builtin_shufflevector(lo, hi, 0, 1, 2, 3, 4, 5, 6, 7,
                                         8, 9, 10, 11, 12, 13, 14, 15);
}

// ---------------------------------------------------------------------------
// Kernel 1: fp32 -> bf16 conversion
// ---------------------------------------------------------------------------
__global__ void mha_cvt_bf16(const float* __restrict__ src,
                             __bf16* __restrict__ dst, int n) {
  int i = blockIdx.x * blockDim.x + threadIdx.x;
  int stride = gridDim.x * blockDim.x;
  for (; i < n; i += stride) dst[i] = (__bf16)src[i];
}

// ---------------------------------------------------------------------------
// Kernel 2: C[M,1024] = A[M,1024] x W[1024,1024]^T   (NT GEMM)
// 128x128 tile / WG, 8 waves (each 32x64), kstep=32, double-buffered staging.
// ---------------------------------------------------------------------------
template <bool WF32>
__global__ __launch_bounds__(256) void
mha_gemm_nt(const __bf16* __restrict__ A, const __bf16* __restrict__ W,
            __bf16* __restrict__ outb, float* __restrict__ outf,
            const float* __restrict__ bias) {
  constexpr int K = D_;
#if HAVE_TDM || HAVE_ASYNC_LDS
  __shared__ alignas(16) __bf16 As[2][128 * 32];
  __shared__ alignas(16) __bf16 Bs[2][128 * 32];
#else
  __shared__ alignas(16) __bf16 As[1][128 * 32];
  __shared__ alignas(16) __bf16 Bs[1][128 * 32];
#endif

  const int tid  = threadIdx.x;
  const int wave = tid >> 5;
  const int lane = tid & 31;
  const int wm   = wave & 3;
  const int wn   = wave >> 2;
  const int lr   = lane & 15;
  const int lh   = lane >> 4;

  const int m0 = blockIdx.y * 128;
  const int n0 = blockIdx.x * 128;

  const int ldr = tid >> 2;         // two rows per thread: ldr, ldr+64
  const int ldk = (tid & 3) * 8;    // 16B per slot

  const __bf16* Ag = A + (size_t)(m0 + ldr) * K + ldk;
  const __bf16* Bg = W + (size_t)(n0 + ldr) * K + ldk;

  v8f acc[2][4] = {};

  auto compute = [&](const __bf16* Asb, const __bf16* Bsb) {
    v16bf af[2], bfm[4];
#pragma unroll
    for (int mi = 0; mi < 2; ++mi)
      af[mi] = ldfrag(&Asb[(wm * 32 + mi * 16 + lr) * 32 + lh * 8]);
#pragma unroll
    for (int ni = 0; ni < 4; ++ni)
      bfm[ni] = ldfrag(&Bsb[(wn * 64 + ni * 16 + lr) * 32 + lh * 8]);
#pragma unroll
    for (int mi = 0; mi < 2; ++mi)
#pragma unroll
      for (int ni = 0; ni < 4; ++ni)
        acc[mi][ni] = wmma_bf16(af[mi], bfm[ni], acc[mi][ni]);
  };

#if HAVE_TDM
  // One 128x32 tile = one TDM descriptor; only wave 0 drives the DMA.
  auto stage = [&](int buf, int k0) {
    tdm_load_2d_bf16(A + (size_t)m0 * K + k0,
                     (unsigned int)(uintptr_t)&As[buf][0],
                     K, M_, K, 32, 128);
    tdm_load_2d_bf16(W + (size_t)n0 * K + k0,
                     (unsigned int)(uintptr_t)&Bs[buf][0],
                     K, D_, K, 32, 128);
  };
  if (wave == 0) stage(0, 0);
  for (int k0 = 0; k0 < K; k0 += 32) {
    const int p = (k0 >> 5) & 1;
    if (wave == 0) {
      if (k0 + 32 < K) {
        stage(p ^ 1, k0 + 32);    // overlap next tile's DMA with this math
        wait_tensor_le2();        // tile p's 2 descriptors complete
      } else {
        wait_tensor_0();
      }
    }
    __syncthreads();              // tile p visible to all waves
    compute(As[p], Bs[p]);
    __syncthreads();              // all reads of p done before its re-stage
  }
#elif HAVE_ASYNC_LDS
  auto stage = [&](int buf, int k0) {
    async_ld_b128(&As[buf][ldr * 32 + ldk],        Ag + k0);
    async_ld_b128(&As[buf][(ldr + 64) * 32 + ldk], Ag + (size_t)64 * K + k0);
    async_ld_b128(&Bs[buf][ldr * 32 + ldk],        Bg + k0);
    async_ld_b128(&Bs[buf][(ldr + 64) * 32 + ldk], Bg + (size_t)64 * K + k0);
  };
  stage(0, 0);
  for (int k0 = 0; k0 < K; k0 += 32) {
    const int p = (k0 >> 5) & 1;
    if (k0 + 32 < K) {
      stage(p ^ 1, k0 + 32);
      wait_async_le4();
    } else {
      wait_async_0();
    }
    __syncthreads();
    compute(As[p], Bs[p]);
    __syncthreads();
  }
#else
  for (int k0 = 0; k0 < K; k0 += 32) {
    v8bf a0 = *(const v8bf*)(Ag + k0);
    v8bf a1 = *(const v8bf*)(Ag + (size_t)64 * K + k0);
    v8bf b0 = *(const v8bf*)(Bg + k0);
    v8bf b1 = *(const v8bf*)(Bg + (size_t)64 * K + k0);
    __syncthreads();
    *(v8bf*)&As[0][ldr * 32 + ldk] = a0;
    *(v8bf*)&As[0][(ldr + 64) * 32 + ldk] = a1;
    *(v8bf*)&Bs[0][ldr * 32 + ldk] = b0;
    *(v8bf*)&Bs[0][(ldr + 64) * 32 + ldk] = b1;
    __syncthreads();
    compute(As[0], Bs[0]);
  }
#endif

  // Epilogue (branch-free per instantiation).
#pragma unroll
  for (int mi = 0; mi < 2; ++mi)
#pragma unroll
    for (int ni = 0; ni < 4; ++ni)
#pragma unroll
      for (int v = 0; v < 8; ++v) {
        int row = m0 + wm * 32 + mi * 16 + lh * 8 + v;
        int col = n0 + wn * 64 + ni * 16 + lr;
        float val = acc[mi][ni][v];
        if (WF32)
          outf[(size_t)row * D_ + col] = val + bias[col];
        else
          outb[(size_t)row * D_ + col] = (__bf16)val;
      }
}

// ---------------------------------------------------------------------------
// Kernel 3: flash attention.  grid = (L/128, H, B); 256 threads = 8 waves.
// ---------------------------------------------------------------------------
__global__ __launch_bounds__(256) void
mha_flash_attn(const __bf16* __restrict__ Q, const __bf16* __restrict__ Km,
               const __bf16* __restrict__ V, const int* __restrict__ mask,
               __bf16* __restrict__ O) {
  __shared__ alignas(16) __bf16 Qs[128 * 64];
  __shared__ alignas(16) __bf16 Ks[128 * 64];
  __shared__ alignas(16) __bf16 Vt[64 * 128];       // transposed V tile [d][n]
  __shared__ alignas(16) __bf16 Ps[8 * 16 * 128];   // per-wave P scratch

  const int tid  = threadIdx.x;
  const int wave = tid >> 5;
  const int lane = tid & 31;
  const int lr   = lane & 15;
  const int lh   = lane >> 4;

  const int m0 = blockIdx.x * 128;
  const int h  = blockIdx.y;
  const int b  = blockIdx.z;

  const int qr = tid >> 3;          // 32 rows x 64 cols per pass, 4 passes
  const int qc = (tid & 7) * 8;

#pragma unroll
  for (int p = 0; p < 4; ++p) {
    int r = qr + p * 32;
    size_t grow = (size_t)(b * L_ + m0 + r);
#if HAVE_ASYNC_LDS
    async_ld_b128(&Qs[r * 64 + qc], Q + grow * D_ + h * DH_ + qc);
#else
    *(v8bf*)&Qs[r * 64 + qc] = *(const v8bf*)(Q + grow * D_ + h * DH_ + qc);
#endif
  }

  float mrun[8], lrun[8], alpha[8];
  v8f o[4] = {};
#pragma unroll
  for (int v = 0; v < 8; ++v) { mrun[v] = -3.0e38f; lrun[v] = 0.0f; }

  __bf16* pw = &Ps[wave * 16 * 128];

  for (int nt = 0; nt < L_ / 128; ++nt) {
    const int n0 = nt * 128;
#pragma unroll
    for (int p = 0; p < 4; ++p) {
      int r = qr + p * 32;
      size_t grow = (size_t)(b * L_ + n0 + r);
#if HAVE_ASYNC_LDS
      async_ld_b128(&Ks[r * 64 + qc], Km + grow * D_ + h * DH_ + qc);
#else
      *(v8bf*)&Ks[r * 64 + qc] = *(const v8bf*)(Km + grow * D_ + h * DH_ + qc);
#endif
      v8bf vv = *(const v8bf*)(V + grow * D_ + h * DH_ + qc);
#pragma unroll
      for (int j = 0; j < 8; ++j) Vt[(qc + j) * 128 + r] = vv[j];
      if (nt + 1 < L_ / 128) {
        size_t nrow = (size_t)(b * L_ + n0 + 128 + r);
        __builtin_prefetch(Km + nrow * D_ + h * DH_ + qc, 0, 3);
        __builtin_prefetch(V + nrow * D_ + h * DH_ + qc, 0, 3);
      }
    }
#if HAVE_ASYNC_LDS
    wait_async_0();                 // Q (first iter) + K tile landed in LDS
#endif
    __syncthreads();

    // ---- S = Q * K^T ----
    v8f s[8] = {};
#pragma unroll
    for (int kk = 0; kk < 64; kk += 32) {
      v16bf a = ldfrag(&Qs[(wave * 16 + lr) * 64 + kk + lh * 8]);
#pragma unroll
      for (int ni = 0; ni < 8; ++ni) {
        v16bf bb = ldfrag(&Ks[(ni * 16 + lr) * 64 + kk + lh * 8]);
        s[ni] = wmma_bf16(a, bb, s[ni]);
      }
    }

    // ---- scale + mask + online softmax ----
    const int mrow_base = m0 + wave * 16 + lh * 8;
    const size_t mb = (size_t)b * L_ * L_;
#pragma unroll
    for (int v = 0; v < 8; ++v) {
      const int mrow = mrow_base + v;
      float tmax = -3.0e38f;
#pragma unroll
      for (int ni = 0; ni < 8; ++ni) {
        float sc = s[ni][v] * SCALE_;
        int mk = mask[mb + (size_t)mrow * L_ + (n0 + ni * 16 + lr)];
        sc += (1.0f - (float)mk) * -100000.0f;
        s[ni][v] = sc;
        tmax = fmaxf(tmax, sc);
      }
#pragma unroll
      for (int off = 8; off >= 1; off >>= 1)
        tmax = fmaxf(tmax, __shfl_xor(tmax, off, 32));
      float mnew = fmaxf(mrun[v], tmax);
      alpha[v] = __expf(mrun[v] - mnew);
      mrun[v] = mnew;
      float tsum = 0.0f;
#pragma unroll
      for (int ni = 0; ni < 8; ++ni) {
        float pv = __expf(s[ni][v] - mnew);
        s[ni][v] = pv;
        tsum += pv;
      }
#pragma unroll
      for (int off = 8; off >= 1; off >>= 1)
        tsum += __shfl_xor(tsum, off, 32);
      lrun[v] = lrun[v] * alpha[v] + tsum;
    }
#pragma unroll
    for (int di = 0; di < 4; ++di)
#pragma unroll
      for (int v = 0; v < 8; ++v) o[di][v] *= alpha[v];

    // ---- P -> per-wave LDS (reshape to A-fragment layout) ----
#pragma unroll
    for (int ni = 0; ni < 8; ++ni)
#pragma unroll
      for (int v = 0; v < 8; ++v)
        pw[(lh * 8 + v) * 128 + ni * 16 + lr] = (__bf16)s[ni][v];

    // ---- O += P * V ----
#pragma unroll
    for (int kk = 0; kk < 4; ++kk) {
      int kb = kk * 32;
      v16bf a = ldfrag(&pw[lr * 128 + kb + lh * 8]);
#pragma unroll
      for (int di = 0; di < 4; ++di) {
        v16bf bb = ldfrag(&Vt[(di * 16 + lr) * 128 + kb + lh * 8]);
        o[di] = wmma_bf16(a, bb, o[di]);
      }
    }
    __syncthreads();
  }

  // ---- normalize + store ----
#pragma unroll
  for (int di = 0; di < 4; ++di)
#pragma unroll
    for (int v = 0; v < 8; ++v) {
      size_t row = (size_t)b * L_ + m0 + wave * 16 + lh * 8 + v;
      int col = h * DH_ + di * 16 + lr;
      O[row * D_ + col] = (__bf16)(o[di][v] / lrun[v]);
    }
}

// ---------------------------------------------------------------------------
// Launch
// ---------------------------------------------------------------------------
extern "C" void kernel_launch(void* const* d_in, const int* in_sizes, int n_in,
                              void* d_out, int out_size, void* d_ws,
                              size_t ws_size, hipStream_t stream) {
  const float* keys = (const float*)d_in[0];
  const float* Wk   = (const float*)d_in[1];
  const float* Wq   = (const float*)d_in[2];
  const float* Wv   = (const float*)d_in[3];
  const float* Wo   = (const float*)d_in[4];
  const float* bo   = (const float*)d_in[5];
  const int*   mask = (const int*)d_in[6];
  float* out = (float*)d_out;

  const size_t MB = 1u << 20;
  char* ws = (char*)d_ws;
  __bf16* Xb  = (__bf16*)(ws + 0 * MB);
  __bf16* Wqb = (__bf16*)(ws + 8 * MB);
  __bf16* Wkb = (__bf16*)(ws + 10 * MB);
  __bf16* Wvb = (__bf16*)(ws + 12 * MB);
  __bf16* Wob = (__bf16*)(ws + 14 * MB);
  __bf16* Qb  = (__bf16*)(ws + 16 * MB);
  __bf16* Kb  = (__bf16*)(ws + 24 * MB);
  __bf16* Vb  = (__bf16*)(ws + 32 * MB);
  __bf16* Ob  = (__bf16*)(ws + 40 * MB);

  mha_cvt_bf16<<<2048, 256, 0, stream>>>(keys, Xb, M_ * D_);
  mha_cvt_bf16<<<512, 256, 0, stream>>>(Wq, Wqb, D_ * D_);
  mha_cvt_bf16<<<512, 256, 0, stream>>>(Wk, Wkb, D_ * D_);
  mha_cvt_bf16<<<512, 256, 0, stream>>>(Wv, Wvb, D_ * D_);
  mha_cvt_bf16<<<512, 256, 0, stream>>>(Wo, Wob, D_ * D_);

  dim3 gg(D_ / 128, M_ / 128);
  mha_gemm_nt<false><<<gg, 256, 0, stream>>>(Xb, Wqb, Qb, nullptr, nullptr);
  mha_gemm_nt<false><<<gg, 256, 0, stream>>>(Xb, Wkb, Kb, nullptr, nullptr);
  mha_gemm_nt<false><<<gg, 256, 0, stream>>>(Xb, Wvb, Vb, nullptr, nullptr);

  dim3 ga(L_ / 128, H_, B_);
  mha_flash_attn<<<ga, 256, 0, stream>>>(Qb, Kb, Vb, mask, Ob);

  mha_gemm_nt<true><<<gg, 256, 0, stream>>>(Ob, Wob, nullptr, out, bo);
}
